// M_A_58213986730361
// MI455X (gfx1250) — compile-verified, matrix operand order checked by reading, not run on hardware
//
#include <hip/hip_runtime.h>
#include <hip/hip_bf16.h>

// ---------------------------------------------------------------------------
// Types for CDNA5 WMMA / TDM
// ---------------------------------------------------------------------------
typedef __bf16 bf16_t;
typedef bf16_t v16bf __attribute__((ext_vector_type(16)));
typedef float  v8f   __attribute__((ext_vector_type(8)));
typedef unsigned int v4u __attribute__((ext_vector_type(4)));
typedef int v4i __attribute__((ext_vector_type(4)));
typedef int v8i __attribute__((ext_vector_type(8)));

#if __has_builtin(__builtin_amdgcn_tensor_load_to_lds)
#define USE_TDM 1
#endif

// Model dimensions (from the reference)
#define T_STEPS 100
#define NBATCH  1024
#define D_IN    410
#define D_L     300
#define D_A     74
#define D_V     36

// GEMM tiling: 256 threads = 8 waves (wave32). Block tile 64(M) x 128(N),
// K tile 64 -> 8 v_wmma_f32_16x16x32_bf16 per staging round per wave.
// Double-buffered LDS (48KB/block) so the TDM DMAs tile k+1 while the
// SIMDs run WMMAs on tile k.
#define BM 64
#define BN 128
#define BK2 64

// ---------------------------------------------------------------------------
// TDM 2-D tile load descriptor (D# groups per CDNA5 ISA ch.8) and issue.
// Loads a tile of (tile_d1 rows x tile_d0 bf16 elems) from a row-major
// tensor with row stride stride_elems into LDS at lds_off. OOB -> zero.
// ---------------------------------------------------------------------------
__device__ __forceinline__ void tdm_load_2d(unsigned int lds_off,
                                            const void* gptr,
                                            unsigned int tensor_d0,
                                            unsigned int tensor_d1,
                                            unsigned int tile_d0,
                                            unsigned int tile_d1,
                                            unsigned int stride_elems)
{
#ifdef USE_TDM
    unsigned long long ga = (unsigned long long)(size_t)gptr;
    v4u g0;
    g0[0] = 1u;                                   // count=1, is_restore=0
    g0[1] = lds_off;                              // lds_addr (bytes)
    g0[2] = (unsigned int)ga;                     // global_addr[31:0]
    g0[3] = (unsigned int)((ga >> 32) & 0x01FFFFFFu) | (2u << 30); // addr[56:32]|type=2
    v8i g1;
    g1[0] = (int)(1u << 16);                      // workgroup_mask=0, data_size=1 (2B)
    g1[1] = (int)((tensor_d0 & 0xFFFFu) << 16);   // tensor_dim0[15:0] @ bits63:48
    g1[2] = (int)((tensor_d0 >> 16) | ((tensor_d1 & 0xFFFFu) << 16)); // d0 hi | d1 lo
    g1[3] = (int)((tensor_d1 >> 16) | (tile_d0 << 16));              // d1 hi | tile_dim0
    g1[4] = (int)(tile_d1 & 0xFFFFu);             // tile_dim1 | tile_dim2=0
    g1[5] = (int)stride_elems;                    // tensor_dim0_stride[31:0]
    g1[6] = 0;                                    // stride hi | dim1_stride lo
    g1[7] = 0;                                    // dim1_stride hi
    v4i gz = {0, 0, 0, 0};
#if __clang_major__ >= 23
    v8i gz8 = {0, 0, 0, 0, 0, 0, 0, 0};
    __builtin_amdgcn_tensor_load_to_lds(g0, g1, gz, gz, gz8, 0);
#else
    __builtin_amdgcn_tensor_load_to_lds(g0, g1, gz, gz, 0);
#endif
#endif
}

// ---------------------------------------------------------------------------
// WMMA GEMM:  C[M,N] = act( A[M,K] * B[N,K]^T + bias[N] )
//   A: bf16 row-major, lda = K (K multiple of 64, pre-padded with zeros)
//   B: bf16 row-major (torch Linear weight layout: dout x din_padded)
//   C: f32, row stride ldc ; act: 0=none 1=relu 2=sigmoid 3=tanh
//   Requires M % 64 == 0 (true for all call sites: 1024 / 102400).
// ---------------------------------------------------------------------------
__global__ __launch_bounds__(256)
void wmma_gemm_kernel(const bf16_t* __restrict__ A,
                      const bf16_t* __restrict__ B,
                      const float*  __restrict__ bias,
                      float* __restrict__ C,
                      int M, int N, int K, int ldc, int act)
{
    __shared__ bf16_t As[2][BM][BK2];
    __shared__ bf16_t Bs[2][BN][BK2];

    const int tid  = threadIdx.x;
    const int bm   = blockIdx.y * BM;
    const int bn   = blockIdx.x * BN;
    const int wave = tid >> 5;
    const int lane = tid & 31;
    const int wm   = (wave >> 2) * 32;     // 0 or 32
    const int wn   = (wave & 3) * 32;      // 0,32,64,96
    const int m16  = lane & 15;
    const int kh   = lane >> 4;            // K-half selector per ISA layout

    v8f acc[2][2] = {};

    const int nk = K / BK2;

#ifdef USE_TDM
    // ---- prologue: DMA tile 0 into buffer 0 ----
    if (tid == 0) {
        tdm_load_2d((unsigned int)(size_t)&As[0][0][0],
                    A + (size_t)bm * K,
                    (unsigned int)K, (unsigned int)(M - bm),
                    BK2, BM, (unsigned int)K);
        tdm_load_2d((unsigned int)(size_t)&Bs[0][0][0],
                    B + (size_t)bn * K,
                    (unsigned int)K, (unsigned int)(N - bn),
                    BK2, BN, (unsigned int)K);
    }
#endif

    for (int kt = 0; kt < nk; ++kt) {
        const int cur = kt & 1;
        const int k0  = kt * BK2;

#ifdef USE_TDM
        // ---- pipelined staging: DMA tile kt+1 into the alternate buffer,
        //      then wait for tile kt's pair (in-order completion) ----
        if (tid == 0) {
            if (kt + 1 < nk) {
                const int k1 = k0 + BK2;
                tdm_load_2d((unsigned int)(size_t)&As[cur ^ 1][0][0],
                            A + (size_t)bm * K + k1,
                            (unsigned int)(K - k1), (unsigned int)(M - bm),
                            BK2, BM, (unsigned int)K);
                tdm_load_2d((unsigned int)(size_t)&Bs[cur ^ 1][0][0],
                            B + (size_t)bn * K + k1,
                            (unsigned int)(K - k1), (unsigned int)(N - bn),
                            BK2, BN, (unsigned int)K);
                __builtin_amdgcn_s_wait_tensorcnt(2);
            } else {
                __builtin_amdgcn_s_wait_tensorcnt(0);
            }
        }
        __syncthreads();
#else
        // ---- fallback: coalesced b128 staging into buffer `cur` ----
        {
            const int arow = tid >> 3;             // 0..31
            const int acol = (tid & 7) * 8;        // 0..56 (16B granules)
#pragma unroll
            for (int rr = 0; rr < 2; ++rr) {
                const int r = arow + rr * 32;
                uint4 av = make_uint4(0u, 0u, 0u, 0u);
                if (bm + r < M)
                    av = *(const uint4*)(A + (size_t)(bm + r) * K + k0 + acol);
                *(uint4*)&As[cur][r][acol] = av;
            }
#pragma unroll
            for (int rr = 0; rr < 4; ++rr) {
                const int r = arow + rr * 32;
                uint4 bv = make_uint4(0u, 0u, 0u, 0u);
                if (bn + r < N)
                    bv = *(const uint4*)(B + (size_t)(bn + r) * K + k0 + acol);
                *(uint4*)&Bs[cur][r][acol] = bv;
            }
            if (k0 + BK2 < K && bm + arow < M)
                __builtin_prefetch(A + (size_t)(bm + arow) * K + k0 + BK2 + acol, 0, 0);
        }
        __syncthreads();
#endif

        // ---- 2 K-subtiles x 2x2 WMMA tiles = 8 WMMAs per staging round ----
#pragma unroll
        for (int kk = 0; kk < BK2; kk += 32) {
            v16bf afrag[2], bfrag[2];
#pragma unroll
            for (int i = 0; i < 2; ++i) {
                const int r = wm + i * 16 + m16;
                const int c = wn + i * 16 + m16;
#pragma unroll
                for (int v = 0; v < 8; ++v) {
                    const int kb = kk + ((v < 4) ? (8 * kh + 2 * v)
                                                 : (16 + 8 * kh + 2 * (v - 4)));
                    afrag[i][2 * v]     = As[cur][r][kb];
                    afrag[i][2 * v + 1] = As[cur][r][kb + 1];
                    bfrag[i][2 * v]     = Bs[cur][c][kb];
                    bfrag[i][2 * v + 1] = Bs[cur][c][kb + 1];
                }
            }
#pragma unroll
            for (int i = 0; i < 2; ++i)
#pragma unroll
                for (int j = 0; j < 2; ++j)
                    acc[i][j] = __builtin_amdgcn_wmma_f32_16x16x32_bf16(
                        false, afrag[i], false, bfrag[j],
                        (short)0, acc[i][j], false, false);
        }
        __syncthreads();
    }

    // ---- epilogue: bias + activation + guarded store ----
#pragma unroll
    for (int j = 0; j < 2; ++j) {
        const int n = bn + wn + j * 16 + m16;
        if (n >= N) continue;
        const float bv = bias ? bias[n] : 0.0f;
#pragma unroll
        for (int i = 0; i < 2; ++i) {
#pragma unroll
            for (int r = 0; r < 8; ++r) {
                const int m = bm + wm + i * 16 + r + 8 * kh;
                if (m >= M) continue;
                float v = acc[i][j][r] + bv;
                if (act == 1)      v = fmaxf(v, 0.0f);
                else if (act == 2) v = 1.0f / (1.0f + __expf(-v));
                else if (act == 3) v = tanhf(v);
                C[(size_t)m * ldc + n] = v;
            }
        }
    }
}

// ---------------------------------------------------------------------------
// f32 -> bf16 with zero padding of columns [cols, dstStride)
// ---------------------------------------------------------------------------
__global__ void pad_convert_kernel(const float* __restrict__ src, int srcStride,
                                   int cols, bf16_t* __restrict__ dst,
                                   int dstStride, int rows)
{
    int idx = blockIdx.x * blockDim.x + threadIdx.x;
    int total = rows * dstStride;
    if (idx >= total) return;
    int r = idx / dstStride, c = idx - r * dstStride;
    dst[idx] = (c < cols) ? (bf16_t)src[(size_t)r * srcStride + c] : (bf16_t)0.0f;
}

// f32 -> bf16 into a column window (no zero fill)
__global__ void convert_into_kernel(const float* __restrict__ src, int srcStride,
                                    bf16_t* __restrict__ dst, int dstStride,
                                    int colOffset, int rows, int cols)
{
    int idx = blockIdx.x * blockDim.x + threadIdx.x;
    if (idx >= rows * cols) return;
    int r = idx / cols, c = idx - r * cols;
    dst[(size_t)r * dstStride + colOffset + c] = (bf16_t)src[(size_t)r * srcStride + c];
}

// strided f32 2-D copy into a column window of dst
__global__ void copy2d_kernel(const float* __restrict__ src, int srcStride,
                              float* __restrict__ dst, int dstStride,
                              int dstOffset, int rows, int cols)
{
    int idx = blockIdx.x * blockDim.x + threadIdx.x;
    if (idx >= rows * cols) return;
    int r = idx / cols, c = idx - r * cols;
    dst[(size_t)r * dstStride + dstOffset + c] = src[(size_t)r * srcStride + c];
}

__global__ void fill_kernel(float* __restrict__ p, float v, int n)
{
    int idx = blockIdx.x * blockDim.x + threadIdx.x;
    if (idx < n) p[idx] = v;
}

// ---------------------------------------------------------------------------
// LSTM gate fusion: z = zih + zhh (each [n,4H], gates i|f|g|o blocked by H)
// c <- sig(f)*c + sig(i)*tanh(g);  h <- sig(o)*tanh(c)
// ---------------------------------------------------------------------------
__global__ void lstm_gates_kernel(const float* __restrict__ zih,
                                  const float* __restrict__ zhh,
                                  float* __restrict__ c,
                                  float* __restrict__ h_out,
                                  int n, int H)
{
    int idx = blockIdx.x * blockDim.x + threadIdx.x;
    if (idx >= n * H) return;
    int row = idx / H, col = idx - row * H;
    const float* zi = zih + (size_t)row * 4 * H;
    const float* zh = zhh + (size_t)row * 4 * H;
    float gi = zi[col]         + zh[col];
    float gf = zi[H + col]     + zh[H + col];
    float gg = zi[2 * H + col] + zh[2 * H + col];
    float go = zi[3 * H + col] + zh[3 * H + col];
    float si = 1.0f / (1.0f + __expf(-gi));
    float sf = 1.0f / (1.0f + __expf(-gf));
    float so = 1.0f / (1.0f + __expf(-go));
    float c2 = sf * c[idx] + si * tanhf(gg);
    c[idx]     = c2;
    h_out[idx] = so * tanhf(c2);
}

// row-wise softmax (one block per row)
__global__ void softmax_rows_kernel(float* __restrict__ data, int width)
{
    __shared__ float red[256];
    const int row = blockIdx.x;
    float* p = data + (size_t)row * width;
    const int tid = threadIdx.x;

    float mx = -3.0e38f;
    for (int c = tid; c < width; c += 256) mx = fmaxf(mx, p[c]);
    red[tid] = mx; __syncthreads();
    for (int s = 128; s > 0; s >>= 1) {
        if (tid < s) red[tid] = fmaxf(red[tid], red[tid + s]);
        __syncthreads();
    }
    mx = red[0]; __syncthreads();

    float sum = 0.0f;
    for (int c = tid; c < width; c += 256) sum += __expf(p[c] - mx);
    red[tid] = sum; __syncthreads();
    for (int s = 128; s > 0; s >>= 1) {
        if (tid < s) red[tid] += red[tid + s];
        __syncthreads();
    }
    const float inv = 1.0f / red[0];
    for (int c = tid; c < width; c += 256) p[c] = __expf(p[c] - mx) * inv;
}

__global__ void ew_mul_kernel(const float* __restrict__ a,
                              const float* __restrict__ b,
                              float* __restrict__ o, int n)
{
    int idx = blockIdx.x * blockDim.x + threadIdx.x;
    if (idx < n) o[idx] = a[idx] * b[idx];
}

// mem <- g1*mem + g2*cHat  (g1,g2 pre-sigmoided, cHat pre-tanh'd in GEMM)
__global__ void mem_update_kernel(const float* __restrict__ g1,
                                  const float* __restrict__ g2,
                                  float* __restrict__ mem,
                                  const float* __restrict__ cHat, int n)
{
    int idx = blockIdx.x * blockDim.x + threadIdx.x;
    if (idx < n) mem[idx] = g1[idx] * mem[idx] + g2[idx] * cHat[idx];
}

// ---------------------------------------------------------------------------
// MMD pairwise kernel term: accum += scale * mean_ij exp(-||Xi-Yj||^2/dim^2)
// ---------------------------------------------------------------------------
__global__ void mmd_kernel(const float* __restrict__ X,
                           const float* __restrict__ Y,
                           int n, int dim, float scale,
                           float* __restrict__ accum)
{
    __shared__ float xs[128];
    __shared__ float red[256];
    const int i = blockIdx.x;
    for (int d = threadIdx.x; d < dim; d += blockDim.x)
        xs[d] = X[(size_t)i * dim + d];
    __syncthreads();

    const float invd2 = 1.0f / ((float)dim * (float)dim);
    float s = 0.0f;
    for (int j = threadIdx.x; j < n; j += blockDim.x) {
        const float* y = Y + (size_t)j * dim;
        float acc = 0.0f;
        for (int d = 0; d < dim; ++d) {
            float df = xs[d] - y[d];
            acc += df * df;
        }
        s += __expf(-acc * invd2);
    }
    red[threadIdx.x] = s; __syncthreads();
    for (int st = 128; st > 0; st >>= 1) {
        if (threadIdx.x < st) red[threadIdx.x] += red[threadIdx.x + st];
        __syncthreads();
    }
    if (threadIdx.x == 0)
        atomicAdd(accum, scale * red[0] / ((float)n * (float)n));
}

// ---------------------------------------------------------------------------
// Host orchestration
// ---------------------------------------------------------------------------
extern "C" void kernel_launch(void* const* d_in, const int* in_sizes, int n_in,
                              void* d_out, int out_size, void* d_ws, size_t ws_size,
                              hipStream_t stream)
{
    (void)in_sizes; (void)n_in; (void)out_size; (void)ws_size;
    const int N = NBATCH;

    const float* x = (const float*)d_in[0];
    auto P = [&](int i) { return (const float*)d_in[i]; };

    // input indices (dict flatten order of setup_inputs)
    enum {
        I_ENC_WIH = 1, I_ENC_WHH, I_ENC_BIH, I_ENC_BHH,
        I_ENCFC_W, I_ENCFC_B,
        I_L_WIH, I_L_WHH, I_L_BIH, I_L_BHH,
        I_A_WIH, I_A_WHH, I_A_BIH, I_A_BHH,
        I_V_WIH, I_V_WHH, I_V_BIH, I_V_BHH,
        I_AT1W1, I_AT1B1, I_AT1W2, I_AT1B2,
        I_AT2W1, I_AT2B1, I_AT2W2, I_AT2B2,
        I_G1W1, I_G1B1, I_G1W2, I_G1B2,
        I_G2W1, I_G2B1, I_G2W2, I_G2B2,
        I_L2ZY_W, I_L2ZY_B,
        I_ZYFY1_W, I_ZYFY1_B, I_ZYFY2_W, I_ZYFY2_B,
        I_ZLFL1_W, I_ZLFL1_B, I_ZLFL2_W, I_ZLFL2_B,
        I_FYY1_W, I_FYY1_B, I_FYY2_W, I_FYY2_B,
        I_DL_WIH, I_DL_WHH, I_DL_BIH, I_DL_BHH, I_DLFC_W, I_DLFC_B,
        I_DA_WIH, I_DA_WHH, I_DA_BIH, I_DA_BHH, I_DAFC_W, I_DAFC_B,
        I_DV_WIH, I_DV_WHH, I_DV_BIH, I_DV_BHH, I_DVFC_W, I_DVFC_B,
        I_NOISE_ZL, I_NOISE_ZY
    };

    // workspace bump allocator
    char* wsp = (char*)d_ws;
    size_t used = 0;
    auto alloc = [&](size_t bytes) -> void* {
        void* p = wsp + used;
        used = (used + bytes + 255) & ~(size_t)255;
        return p;
    };

    auto padcvt = [&](const float* src, int srcStride, int cols,
                      bf16_t* dst, int dstStride, int rows) {
        int total = rows * dstStride;
        pad_convert_kernel<<<(total + 255) / 256, 256, 0, stream>>>(
            src, srcStride, cols, dst, dstStride, rows);
    };
    auto cvt_into = [&](const float* src, int srcStride, bf16_t* dst,
                        int dstStride, int colOff, int rows, int cols) {
        int total = rows * cols;
        convert_into_kernel<<<(total + 255) / 256, 256, 0, stream>>>(
            src, srcStride, dst, dstStride, colOff, rows, cols);
    };
    auto gemm = [&](const bf16_t* A, const bf16_t* Bw, const float* bias,
                    float* C, int M, int Nn, int K, int ldc, int act) {
        dim3 g((Nn + BN - 1) / BN, (M + BM - 1) / BM);
        wmma_gemm_kernel<<<g, 256, 0, stream>>>(A, Bw, bias, C, M, Nn, K, ldc, act);
    };
    auto copy2d = [&](const float* src, int ss, float* dst, int ds, int off,
                      int rows, int cols) {
        int total = rows * cols;
        copy2d_kernel<<<(total + 255) / 256, 256, 0, stream>>>(
            src, ss, dst, ds, off, rows, cols);
    };
    auto zero = [&](float* p, int n) {
        fill_kernel<<<(n + 255) / 256, 256, 0, stream>>>(p, 0.0f, n);
    };
    auto gates = [&](const float* zih, const float* zhh, float* c, float* h,
                     int H) {
        int total = N * H;
        lstm_gates_kernel<<<(total + 255) / 256, 256, 0, stream>>>(
            zih, zhh, c, h, N, H);
    };
    auto cvtW = [&](int widx, int dout, int din, int Kpad) -> bf16_t* {
        bf16_t* p = (bf16_t*)alloc((size_t)dout * Kpad * sizeof(bf16_t));
        padcvt(P(widx), din, din, p, Kpad, dout);
        return p;
    };

    // ---- weights -> padded bf16 (all K multiples of 64) ----
    bf16_t* Wenc_ih = cvtW(I_ENC_WIH, 512, D_IN, 448);
    bf16_t* Wenc_hh = cvtW(I_ENC_WHH, 512, 128, 128);
    bf16_t* Wencfc  = cvtW(I_ENCFC_W, 128, 128, 128);
    bf16_t* Wl_ih   = cvtW(I_L_WIH, 512, D_L, 320);
    bf16_t* Wl_hh   = cvtW(I_L_WHH, 512, 128, 128);
    bf16_t* Wa_ih   = cvtW(I_A_WIH, 256, D_A, 128);
    bf16_t* Wa_hh   = cvtW(I_A_WHH, 256, 64, 64);
    bf16_t* Wv_ih   = cvtW(I_V_WIH, 256, D_V, 64);
    bf16_t* Wv_hh   = cvtW(I_V_WHH, 256, 64, 64);
    bf16_t* Wat1a   = cvtW(I_AT1W1, 256, 512, 512);
    bf16_t* Wat1b   = cvtW(I_AT1W2, 512, 256, 256);
    bf16_t* Wat2a   = cvtW(I_AT2W1, 256, 512, 512);
    bf16_t* Wat2b   = cvtW(I_AT2W2, 256, 256, 256);
    bf16_t* Wg1a    = cvtW(I_G1W1, 256, 768, 768);
    bf16_t* Wg1b    = cvtW(I_G1W2, 256, 256, 256);
    bf16_t* Wg2a    = cvtW(I_G2W1, 256, 768, 768);
    bf16_t* Wg2b    = cvtW(I_G2W2, 256, 256, 256);
    bf16_t* Wl2zy   = cvtW(I_L2ZY_W, 128, 512, 512);
    bf16_t* Wzyfy1  = cvtW(I_ZYFY1_W, 128, 128, 128);
    bf16_t* Wzyfy2  = cvtW(I_ZYFY2_W, 128, 128, 128);
    bf16_t* Wzlfl1  = cvtW(I_ZLFL1_W, 128, 128, 128);
    bf16_t* Wzlfl2  = cvtW(I_ZLFL2_W, 128, 128, 128);
    bf16_t* Wfyy1   = cvtW(I_FYY1_W, 128, 128, 128);
    bf16_t* Wfyy2   = cvtW(I_FYY2_W, 1, 128, 128);
    bf16_t* Wdec_ih[3] = { cvtW(I_DL_WIH, 1024, 256, 256),
                           cvtW(I_DA_WIH, 1024, 256, 256),
                           cvtW(I_DV_WIH, 1024, 256, 256) };
    bf16_t* Wdec_hh[3] = { cvtW(I_DL_WHH, 1024, 256, 256),
                           cvtW(I_DA_WHH, 1024, 256, 256),
                           cvtW(I_DV_WHH, 1024, 256, 256) };
    bf16_t* Wdec_fc[3] = { cvtW(I_DLFC_W, D_L, 256, 256),
                           cvtW(I_DAFC_W, D_A, 256, 256),
                           cvtW(I_DVFC_W, D_V, 256, 256) };

    // ---- activation scratch ----
    bf16_t* xt_bf   = (bf16_t*)alloc((size_t)N * 448 * 2);
    bf16_t* h_bf    = (bf16_t*)alloc((size_t)N * 256 * 2);
    bf16_t* wide_bf = (bf16_t*)alloc((size_t)N * 768 * 2);  // cStar/attended/both
    bf16_t* hid_bf  = (bf16_t*)alloc((size_t)N * 256 * 2);
    float* zih   = (float*)alloc((size_t)N * 1024 * 4);
    float* zhh   = (float*)alloc((size_t)N * 1024 * 4);
    float* cStar = (float*)alloc((size_t)N * 512 * 4);
    float* attnd = (float*)alloc((size_t)N * 512 * 4);
    float* t1    = (float*)alloc((size_t)N * 256 * 4);
    float* cHat  = (float*)alloc((size_t)N * 256 * 4);
    float* g1b   = (float*)alloc((size_t)N * 256 * 4);
    float* g2b   = (float*)alloc((size_t)N * 256 * 4);
    float* h_enc = (float*)alloc((size_t)N * 128 * 4);
    float* c_enc = (float*)alloc((size_t)N * 128 * 4);
    float* h_l   = (float*)alloc((size_t)N * 128 * 4);
    float* c_l   = (float*)alloc((size_t)N * 128 * 4);
    float* h_a   = (float*)alloc((size_t)N * 64 * 4);
    float* c_a   = (float*)alloc((size_t)N * 64 * 4);
    float* h_v   = (float*)alloc((size_t)N * 64 * 4);
    float* c_v   = (float*)alloc((size_t)N * 64 * 4);
    float* mem   = (float*)alloc((size_t)N * 256 * 4);
    float* zl    = (float*)alloc((size_t)N * 128 * 4);
    float* zy    = (float*)alloc((size_t)N * 128 * 4);
    float* fy    = (float*)alloc((size_t)N * 128 * 4);
    float* fl    = (float*)alloc((size_t)N * 128 * 4);
    float* fyfl  = (float*)alloc((size_t)N * 256 * 4);
    float* cdec  = (float*)alloc((size_t)N * 256 * 4);
    float* zbuf  = (float*)alloc((size_t)N * 256 * 4);
    float* hs    = (float*)alloc((size_t)T_STEPS * N * 256 * 4);
    bf16_t* hs_bf = (bf16_t*)alloc((size_t)T_STEPS * N * 256 * 2);

    // output layout
    float* out = (float*)d_out;
    const size_t offL = 0;
    const size_t offA = offL + (size_t)T_STEPS * N * D_L;
    const size_t offV = offA + (size_t)T_STEPS * N * D_A;
    const size_t offY = offV + (size_t)T_STEPS * N * D_V;
    const size_t offM = offY + (size_t)N * 1;

    // =======================================================================
    // Encoder LSTM over T
    // =======================================================================
    zero(h_enc, N * 128); zero(c_enc, N * 128);
    for (int t = 0; t < T_STEPS; ++t) {
        const float* xt = x + (size_t)t * N * D_IN;
        padcvt(xt, D_IN, D_IN, xt_bf, 448, N);
        gemm(xt_bf, Wenc_ih, P(I_ENC_BIH), zih, N, 512, 448, 512, 0);
        padcvt(h_enc, 128, 128, h_bf, 128, N);
        gemm(h_bf, Wenc_hh, P(I_ENC_BHH), zhh, N, 512, 128, 512, 0);
        gates(zih, zhh, c_enc, h_enc, 128);
    }
    padcvt(h_enc, 128, 128, h_bf, 128, N);
    gemm(h_bf, Wencfc, P(I_ENCFC_B), zl, N, 128, 128, 128, 0);

    // =======================================================================
    // MFN over T
    // =======================================================================
    zero(h_l, N * 128); zero(c_l, N * 128);
    zero(h_a, N * 64);  zero(c_a, N * 64);
    zero(h_v, N * 64);  zero(c_v, N * 64);
    zero(mem, N * 256);
    for (int t = 0; t < T_STEPS; ++t) {
        const float* xt = x + (size_t)t * N * D_IN;
        // prev cells into cStar[:, 0:256]
        copy2d(c_l, 128, cStar, 512, 0,   N, 128);
        copy2d(c_a, 64,  cStar, 512, 128, N, 64);
        copy2d(c_v, 64,  cStar, 512, 192, N, 64);
        // three modality LSTM cells
        padcvt(xt, D_IN, D_L, xt_bf, 320, N);
        gemm(xt_bf, Wl_ih, P(I_L_BIH), zih, N, 512, 320, 512, 0);
        padcvt(h_l, 128, 128, h_bf, 128, N);
        gemm(h_bf, Wl_hh, P(I_L_BHH), zhh, N, 512, 128, 512, 0);
        gates(zih, zhh, c_l, h_l, 128);

        padcvt(xt + D_L, D_IN, D_A, xt_bf, 128, N);
        gemm(xt_bf, Wa_ih, P(I_A_BIH), zih, N, 256, 128, 256, 0);
        padcvt(h_a, 64, 64, h_bf, 64, N);
        gemm(h_bf, Wa_hh, P(I_A_BHH), zhh, N, 256, 64, 256, 0);
        gates(zih, zhh, c_a, h_a, 64);

        padcvt(xt + D_L + D_A, D_IN, D_V, xt_bf, 64, N);
        gemm(xt_bf, Wv_ih, P(I_V_BIH), zih, N, 256, 64, 256, 0);
        padcvt(h_v, 64, 64, h_bf, 64, N);
        gemm(h_bf, Wv_hh, P(I_V_BHH), zhh, N, 256, 64, 256, 0);
        gates(zih, zhh, c_v, h_v, 64);

        // new cells into cStar[:, 256:512]
        copy2d(c_l, 128, cStar, 512, 256, N, 128);
        copy2d(c_a, 64,  cStar, 512, 384, N, 64);
        copy2d(c_v, 64,  cStar, 512, 448, N, 64);

        // attention 1 -> softmax -> attended
        padcvt(cStar, 512, 512, wide_bf, 512, N);
        gemm(wide_bf, Wat1a, P(I_AT1B1), t1, N, 256, 512, 256, 1);
        padcvt(t1, 256, 256, hid_bf, 256, N);
        gemm(hid_bf, Wat1b, P(I_AT1B2), attnd, N, 512, 256, 512, 0);
        softmax_rows_kernel<<<N, 256, 0, stream>>>(attnd, 512);
        ew_mul_kernel<<<(N * 512 + 255) / 256, 256, 0, stream>>>(
            attnd, cStar, attnd, N * 512);

        // attention 2 -> cHat (tanh fused)
        padcvt(attnd, 512, 512, wide_bf, 512, N);
        gemm(wide_bf, Wat2a, P(I_AT2B1), t1, N, 256, 512, 256, 1);
        padcvt(t1, 256, 256, hid_bf, 256, N);
        gemm(hid_bf, Wat2b, P(I_AT2B2), cHat, N, 256, 256, 256, 3);

        // both = [attended, mem] in bf16 (768 wide)
        padcvt(attnd, 512, 512, wide_bf, 768, N);     // zero-fills 512..767
        cvt_into(mem, 256, wide_bf, 768, 512, N, 256);

        // gamma1 / gamma2 (sigmoid fused)
        gemm(wide_bf, Wg1a, P(I_G1B1), t1, N, 256, 768, 256, 1);
        padcvt(t1, 256, 256, hid_bf, 256, N);
        gemm(hid_bf, Wg1b, P(I_G1B2), g1b, N, 256, 256, 256, 2);
        gemm(wide_bf, Wg2a, P(I_G2B1), t1, N, 256, 768, 256, 1);
        padcvt(t1, 256, 256, hid_bf, 256, N);
        gemm(hid_bf, Wg2b, P(I_G2B2), g2b, N, 256, 256, 256, 2);

        mem_update_kernel<<<(N * 256 + 255) / 256, 256, 0, stream>>>(
            g1b, g2b, mem, cHat, N * 256);
    }

    // mfn_last = [h_l, h_a, h_v, mem] -> zy
    copy2d(h_l, 128, cStar, 512, 0,   N, 128);
    copy2d(h_a, 64,  cStar, 512, 128, N, 64);
    copy2d(h_v, 64,  cStar, 512, 192, N, 64);
    copy2d(mem, 256, cStar, 512, 256, N, 256);
    padcvt(cStar, 512, 512, wide_bf, 512, N);
    gemm(wide_bf, Wl2zy, P(I_L2ZY_B), zy, N, 128, 512, 128, 0);

    // =======================================================================
    // MMD loss into out[offM]
    // =======================================================================
    zero(out + offM, 1);
    const float* nzl = P(I_NOISE_ZL);
    const float* nzy = P(I_NOISE_ZY);
    mmd_kernel<<<N, 256, 0, stream>>>(nzl, nzl, N, 128,  1.0f, out + offM);
    mmd_kernel<<<N, 256, 0, stream>>>(zl,  zl,  N, 128,  1.0f, out + offM);
    mmd_kernel<<<N, 256, 0, stream>>>(nzl, zl,  N, 128, -2.0f, out + offM);
    mmd_kernel<<<N, 256, 0, stream>>>(nzy, nzy, N, 128,  1.0f, out + offM);
    mmd_kernel<<<N, 256, 0, stream>>>(zy,  zy,  N, 128,  1.0f, out + offM);
    mmd_kernel<<<N, 256, 0, stream>>>(nzy, zy,  N, 128, -2.0f, out + offM);

    // =======================================================================
    // fy / fl heads, fyfl
    // =======================================================================
    padcvt(zy, 128, 128, h_bf, 128, N);
    gemm(h_bf, Wzyfy1, P(I_ZYFY1_B), t1, N, 128, 128, 128, 1);
    padcvt(t1, 128, 128, h_bf, 128, N);
    gemm(h_bf, Wzyfy2, P(I_ZYFY2_B), fy, N, 128, 128, 128, 1);

    padcvt(zl, 128, 128, h_bf, 128, N);
    gemm(h_bf, Wzlfl1, P(I_ZLFL1_B), t1, N, 128, 128, 128, 1);
    padcvt(t1, 128, 128, h_bf, 128, N);
    gemm(h_bf, Wzlfl2, P(I_ZLFL2_B), fl, N, 128, 128, 128, 1);

    copy2d(fy, 128, fyfl, 256, 0,   N, 128);
    copy2d(fl, 128, fyfl, 256, 128, N, 128);

    // y_hat = linear(relu(linear(fy)))
    padcvt(fy, 128, 128, h_bf, 128, N);
    gemm(h_bf, Wfyy1, P(I_FYY1_B), t1, N, 128, 128, 128, 1);
    padcvt(t1, 128, 128, h_bf, 128, N);
    gemm(h_bf, Wfyy2, P(I_FYY2_B), out + offY, N, 1, 128, 1, 0);

    // =======================================================================
    // Three decoders
    // =======================================================================
    const float* dec_bih[3] = { P(I_DL_BIH), P(I_DA_BIH), P(I_DV_BIH) };
    const float* dec_bhh[3] = { P(I_DL_BHH), P(I_DA_BHH), P(I_DV_BHH) };
    const float* dec_fcb[3] = { P(I_DLFC_B), P(I_DAFC_B), P(I_DVFC_B) };
    const int    dec_dout[3] = { D_L, D_A, D_V };
    const size_t dec_off[3]  = { offL, offA, offV };

    zero(zbuf, N * 256);
    for (int d = 0; d < 3; ++d) {
        zero(cdec, N * 256);
        const float* inp = fyfl;
        const float* hprev = zbuf;
        for (int t = 0; t < T_STEPS; ++t) {
            padcvt(inp, 256, 256, hid_bf, 256, N);
            gemm(hid_bf, Wdec_ih[d], dec_bih[d], zih, N, 1024, 256, 1024, 0);
            padcvt(hprev, 256, 256, h_bf, 256, N);
            gemm(h_bf, Wdec_hh[d], dec_bhh[d], zhh, N, 1024, 256, 1024, 0);
            float* hcur = hs + (size_t)t * N * 256;
            gates(zih, zhh, cdec, hcur, 256);
            inp = hcur;
            hprev = hcur;
        }
        // batched output projection over all T at once
        padcvt(hs, 256, 256, hs_bf, 256, T_STEPS * N);
        gemm(hs_bf, Wdec_fc[d], dec_fcb[d], out + dec_off[d],
             T_STEPS * N, dec_dout[d], 256, dec_dout[d], 0);
    }
}
